// MinGRUStack_13185549599204
// MI455X (gfx1250) — matrix operand order repaired and checked by reading, not compiled, not run on hardware
//
#include <hip/hip_runtime.h>
#include <hip/hip_bf16.h>
#include <math.h>

// Problem constants (match reference)
#define B_   8
#define T_   2048
#define D_   512
#define P_   8
#define N_   (B_ * T_)        // 16384 rows
#define KDIM (D_ * P_)        // 4096 GEMM K dimension
#define KITER (KDIM / 32)     // 128 K steps
#define CH_  16               // recurrence chunks
#define LCH_ (T_ / CH_)       // 128 steps per chunk
#define NCHAIN (B_ * D_)      // 4096 scan chains

typedef _Float16 h16;
typedef h16  v16h __attribute__((ext_vector_type(16)));
typedef h16  v8h  __attribute__((ext_vector_type(8)));
typedef float v8f __attribute__((ext_vector_type(8)));
typedef int  v4i_ __attribute__((ext_vector_type(4)));

// ---- CDNA5 async global->LDS copy (ASYNCcnt-tracked), with safe fallback ----
__device__ __forceinline__ void async_copy_b128(const h16* src, h16* dst) {
#if __has_builtin(__builtin_amdgcn_global_load_async_to_lds_b128)
    __builtin_amdgcn_global_load_async_to_lds_b128(
        (__attribute__((address_space(1))) v4i_*)src,
        (__attribute__((address_space(3))) v4i_*)dst, 0, 0);
#else
    *(v8h*)dst = *(const v8h*)src;
#endif
}
__device__ __forceinline__ void wait_async0() {
#if __has_builtin(__builtin_amdgcn_global_load_async_to_lds_b128)
#if __has_builtin(__builtin_amdgcn_s_wait_asynccnt)
    __builtin_amdgcn_s_wait_asynccnt(0);
#else
    asm volatile("s_wait_asynccnt 0" ::: "memory");
#endif
#endif
}
__device__ __forceinline__ float fast_rcp(float x) {
#if __has_builtin(__builtin_amdgcn_rcpf)
    return __builtin_amdgcn_rcpf(x);
#else
    return 1.0f / x;
#endif
}

// ---------------------------------------------------------------------------
// Convert V (KDIM x D_, f32, row-major) -> Vt (D_ x KDIM, f16, row-major).
// ---------------------------------------------------------------------------
__global__ __launch_bounds__(256)
void cvt_transpose(const float* __restrict__ v, h16* __restrict__ vt) {
    __shared__ float tile[32][33];
    const int k0 = blockIdx.x * 32;
    const int o0 = blockIdx.y * 32;
    const int c  = threadIdx.x & 31;
    const int r4 = threadIdx.x >> 5;
#pragma unroll
    for (int q = 0; q < 4; ++q) {
        int r = r4 * 4 + q;
        tile[r][c] = v[(size_t)(k0 + r) * D_ + o0 + c];
    }
    __syncthreads();
#pragma unroll
    for (int q = 0; q < 4; ++q) {
        int r = r4 * 4 + q;
        vt[(size_t)(o0 + r) * KDIM + k0 + c] = (h16)tile[c][r];
    }
}

// ---------------------------------------------------------------------------
// f32 tiled transpose: src (rows x cols) -> dst (cols x rows). Coalesced both ways.
// ---------------------------------------------------------------------------
__global__ __launch_bounds__(256)
void transpose_f32(const float* __restrict__ src, float* __restrict__ dst,
                   int rows, int cols) {
    __shared__ float tile[32][33];
    const int r0 = blockIdx.x * 32;
    const int c0 = blockIdx.y * 32;
    const int c  = threadIdx.x & 31;
    const int r4 = threadIdx.x >> 5;
#pragma unroll
    for (int q = 0; q < 4; ++q) {
        int r = r4 * 4 + q;
        tile[r][c] = src[(size_t)(r0 + r) * cols + c0 + c];
    }
    __syncthreads();
#pragma unroll
    for (int q = 0; q < 4; ++q) {
        int r = r4 * 4 + q;
        dst[(size_t)(c0 + r) * rows + r0 + c] = tile[c][r];
    }
}

// ---------------------------------------------------------------------------
// Fused APL layer as WMMA GEMM: Y(N_ x D_) = C(N_ x KDIM) * V.
// Coefficients built on the fly from transposed activations Xt (coalesced),
// B tiles streamed with global_load_async_to_lds_b128, double-buffered LDS,
// four v_wmma_f32_16x16x32_f16 per wave per K-step.
// ---------------------------------------------------------------------------
#define BM 128
#define BN 64
#define BK 32
#define LSTR 40   // padded LDS row stride in halves (80B, keeps 16B alignment)

__global__ __launch_bounds__(256)
void apl_gemm(const float* __restrict__ Xt,   // (D_, N_) transposed activations
              const float* __restrict__ Pp,   // (D_, P_) sorted breakpoints
              const h16*  __restrict__ Vt,    // (D_, KDIM) f16, transposed V
              float* __restrict__ Y,          // (N_, D_)
              int do_sigmoid) {
    __shared__ float p_s[D_ * P_];            // 16 KB
    __shared__ h16   Als[2][BM * LSTR];       // 2 x 10 KB (double buffer)
    __shared__ h16   Bls[2][BN * LSTR];       // 2 x  5 KB (double buffer)

    const int t     = threadIdx.x;
    const int lane  = t & 31;
    const int wave  = t >> 5;
    const int waveM = wave >> 1;              // 0..3
    const int waveN = wave & 1;               // 0..1
    const int l15   = lane & 15;
    const int hi    = lane >> 4;              // 0 or 1
    const int row0  = blockIdx.y * BM;
    const int col0  = blockIdx.x * BN;

    for (int i = t; i < D_ * P_; i += 256) p_s[i] = Pp[i];

    // Stage one K-step: builds A coeffs + issues async B copy into buffer `buf`.
    auto stage = [&](int kt, int buf) {
        // B tile: 64 cols x 32 K halves; one async 16B lane-copy per thread.
        {
            int n  = t >> 2;
            int ch = (t & 3) * 8;
            async_copy_b128(&Vt[(size_t)(col0 + n) * KDIM + kt * BK + ch],
                            &Bls[buf][n * LSTR + ch]);
        }
        // A tile: this K-chunk covers 4 input features; coalesced reads of Xt.
#pragma unroll
        for (int q = 0; q < 2; ++q) {
            int pr   = t + q * 256;           // 0..511 (row, feature) pairs
            int r    = pr & (BM - 1);
            int il   = pr >> 7;               // 0..3
            int feat = kt * 4 + il;
            float xv = Xt[(size_t)feat * N_ + row0 + r];
            const float* pi = &p_s[feat * P_];
            int idx = 0;
#pragma unroll
            for (int j = 0; j < P_; ++j) idx += (pi[j] < xv) ? 1 : 0;  // searchsorted
            idx = max(1, min(P_ - 1, idx));
            int k = idx - 1;
            float pl = pi[k], prt = pi[idx];
            float tt = (xv - pl) * fast_rcp(prt - pl);
            tt = fminf(1.0f, fmaxf(0.0f, tt));
            float omt = 1.0f - tt;
            h16 carr[8] __attribute__((aligned(16)));
#pragma unroll
            for (int pp = 0; pp < P_; ++pp) {
                float val = (pp == k) ? omt : ((pp == idx) ? tt : 0.0f);
                carr[pp] = (h16)val;
            }
            *(v8h*)(&Als[buf][r * LSTR + il * 8]) = *(const v8h*)carr;
        }
    };

    v8f acc[2][2] = {};

    stage(0, 0);
    wait_async0();
    __syncthreads();

    for (int kt = 0; kt < KITER; ++kt) {
        const int cur = kt & 1;
        if (kt + 1 < KITER) stage(kt + 1, cur ^ 1);

        // Fragments per CDNA5 16-bit A(16x32) / B(32x16) VGPR layouts.
        v16h af[2], bf[2];
#pragma unroll
        for (int a = 0; a < 2; ++a) {
            const h16* ap = &Als[cur][(waveM * 32 + a * 16 + l15) * LSTR + (hi ? 8 : 0)];
            v8h lo = *(const v8h*)ap;
            v8h hh = *(const v8h*)(ap + 16);
#pragma unroll
            for (int e = 0; e < 8; ++e) { af[a][e] = lo[e]; af[a][e + 8] = hh[e]; }
        }
#pragma unroll
        for (int b = 0; b < 2; ++b) {
            const h16* bp = &Bls[cur][(waveN * 32 + b * 16 + l15) * LSTR + (hi ? 8 : 0)];
            v8h lo = *(const v8h*)bp;
            v8h hh = *(const v8h*)(bp + 16);
#pragma unroll
            for (int e = 0; e < 8; ++e) { bf[b][e] = lo[e]; bf[b][e + 8] = hh[e]; }
        }
#pragma unroll
        for (int a = 0; a < 2; ++a)
#pragma unroll
            for (int b = 0; b < 2; ++b)
                acc[a][b] = __builtin_amdgcn_wmma_f32_16x16x32_f16(
                    false, af[a], false, bf[b], (short)0, acc[a][b], false, false);

        if (kt + 1 < KITER) wait_async0();
        __syncthreads();
    }

    // Epilogue (optional sigmoid for z-gate GEMMs), f32 store per C/D layout.
#pragma unroll
    for (int a = 0; a < 2; ++a)
#pragma unroll
        for (int b = 0; b < 2; ++b) {
            int colg = col0 + waveN * 32 + b * 16 + l15;
#pragma unroll
            for (int r = 0; r < 8; ++r) {
                int rowg = row0 + waveM * 32 + a * 16 + (hi ? r + 8 : r);
                float val = acc[a][b][r];
                if (do_sigmoid) val = 1.0f / (1.0f + __expf(-val));
                Y[(size_t)rowg * D_ + colg] = val;
            }
        }
}

// ---------------------------------------------------------------------------
// Chunked parallel scan for h[t] = (1-z) h[t-1] + z h_bar, h0 = 0.
// ---------------------------------------------------------------------------
__global__ __launch_bounds__(256)
void rec_pass1(const float* __restrict__ z, const float* __restrict__ hb,
               float* __restrict__ Aprod, float* __restrict__ Hend) {
    int g    = blockIdx.x * 256 + threadIdx.x;
    int d    = g & (D_ - 1);
    int rest = g >> 9;
    int b    = rest & (B_ - 1);
    int q    = rest >> 3;
    size_t base = ((size_t)b * T_ + (size_t)q * LCH_) * D_ + d;
    float ap = 1.0f, h = 0.0f;
    for (int j = 0; j < LCH_; ++j) {
        float zz = z[base + (size_t)j * D_];
        float hh = hb[base + (size_t)j * D_];
        float a  = 1.0f - zz;
        ap *= a;
        h = a * h + zz * hh;
    }
    int chain = b * D_ + d;
    Aprod[q * NCHAIN + chain] = ap;
    Hend [q * NCHAIN + chain] = h;
}

__global__ __launch_bounds__(256)
void rec_pass2(const float* __restrict__ Aprod, const float* __restrict__ Hend,
               float* __restrict__ Hstart) {
    int chain = blockIdx.x * 256 + threadIdx.x;
    float h = 0.0f;
    for (int q = 0; q < CH_; ++q) {
        Hstart[q * NCHAIN + chain] = h;
        h = Aprod[q * NCHAIN + chain] * h + Hend[q * NCHAIN + chain];
    }
}

__global__ __launch_bounds__(256)
void rec_pass3(const float* __restrict__ z, const float* __restrict__ hb,
               const float* __restrict__ Hstart, float* __restrict__ H) {
    int g    = blockIdx.x * 256 + threadIdx.x;
    int d    = g & (D_ - 1);
    int rest = g >> 9;
    int b    = rest & (B_ - 1);
    int q    = rest >> 3;
    size_t base = ((size_t)b * T_ + (size_t)q * LCH_) * D_ + d;
    float h = Hstart[q * NCHAIN + b * D_ + d];
    for (int j = 0; j < LCH_; ++j) {
        float zz = z[base + (size_t)j * D_];
        float hh = hb[base + (size_t)j * D_];
        h = (1.0f - zz) * h + zz * hh;
        H[base + (size_t)j * D_] = h;
    }
}

// ---------------------------------------------------------------------------
// In-place max-abs row normalization: one wave32 per (b,t) row.
// ---------------------------------------------------------------------------
__global__ __launch_bounds__(256)
void maxabs_norm(float* __restrict__ H) {
    int wave = threadIdx.x >> 5, lane = threadIdx.x & 31;
    int row  = blockIdx.x * 8 + wave;
    size_t base = (size_t)row * D_;
    float vals[16];
    float m = 0.0f;
#pragma unroll
    for (int w = 0; w < 16; ++w) {
        float v = H[base + w * 32 + lane];
        vals[w] = v;
        m = fmaxf(m, fabsf(v));
    }
#pragma unroll
    for (int off = 16; off > 0; off >>= 1)
        m = fmaxf(m, __shfl_xor(m, off, 32));
    float inv = 1.0f / (m + 1e-6f);
#pragma unroll
    for (int w = 0; w < 16; ++w)
        H[base + w * 32 + lane] = vals[w] * inv;
}

// ---------------------------------------------------------------------------
extern "C" void kernel_launch(void* const* d_in, const int* in_sizes, int n_in,
                              void* d_out, int out_size, void* d_ws, size_t ws_size,
                              hipStream_t stream) {
    (void)in_sizes; (void)n_in; (void)out_size; (void)ws_size;
    const float* x   = (const float*)d_in[0];
    const float* pz0 = (const float*)d_in[1];
    const float* vz0 = (const float*)d_in[2];
    const float* ph0 = (const float*)d_in[3];
    const float* vh0 = (const float*)d_in[4];
    const float* pz1 = (const float*)d_in[5];
    const float* vz1 = (const float*)d_in[6];
    const float* ph1 = (const float*)d_in[7];
    const float* vh1 = (const float*)d_in[8];
    const float* po  = (const float*)d_in[9];
    const float* vo  = (const float*)d_in[10];

    float* out = (float*)d_out;                      // (N_, D_)
    float* h1  = out + (size_t)N_ * D_;
    float* h2  = h1  + (size_t)N_ * D_;

    // Workspace layout
    char* ws = (char*)d_ws;
    const size_t vbytes = (size_t)KDIM * D_ * sizeof(h16);   // 4 MB each
    h16* vz0t = (h16*)ws; ws += vbytes;
    h16* vh0t = (h16*)ws; ws += vbytes;
    h16* vz1t = (h16*)ws; ws += vbytes;
    h16* vh1t = (h16*)ws; ws += vbytes;
    h16* vot  = (h16*)ws; ws += vbytes;
    float* zbuf = (float*)ws; ws += (size_t)N_ * D_ * sizeof(float);
    float* hbar = (float*)ws; ws += (size_t)N_ * D_ * sizeof(float);
    float* Xt   = (float*)ws; ws += (size_t)N_ * D_ * sizeof(float);
    float* Aprod  = (float*)ws; ws += (size_t)CH_ * NCHAIN * sizeof(float);
    float* Hend   = (float*)ws; ws += (size_t)CH_ * NCHAIN * sizeof(float);
    float* Hstart = (float*)ws; ws += (size_t)CH_ * NCHAIN * sizeof(float);

    dim3 tgrid(KDIM / 32, D_ / 32);                  // (128, 16)
    cvt_transpose<<<tgrid, 256, 0, stream>>>(vz0, vz0t);
    cvt_transpose<<<tgrid, 256, 0, stream>>>(vh0, vh0t);
    cvt_transpose<<<tgrid, 256, 0, stream>>>(vz1, vz1t);
    cvt_transpose<<<tgrid, 256, 0, stream>>>(vh1, vh1t);
    cvt_transpose<<<tgrid, 256, 0, stream>>>(vo,  vot);

    dim3 xgrid(N_ / 32, D_ / 32);                    // (512, 16)
    dim3 ggrid(D_ / BN, N_ / BM);                    // (8, 128)

    // ---- Layer 1
    transpose_f32<<<xgrid, 256, 0, stream>>>(x, Xt, N_, D_);
    apl_gemm<<<ggrid, 256, 0, stream>>>(Xt, ph0, vh0t, hbar, 0);
    apl_gemm<<<ggrid, 256, 0, stream>>>(Xt, pz0, vz0t, zbuf, 1);
    rec_pass1<<<256, 256, 0, stream>>>(zbuf, hbar, Aprod, Hend);
    rec_pass2<<<16, 256, 0, stream>>>(Aprod, Hend, Hstart);
    rec_pass3<<<256, 256, 0, stream>>>(zbuf, hbar, Hstart, h1);
    maxabs_norm<<<N_ / 8, 256, 0, stream>>>(h1);

    // ---- Layer 2
    transpose_f32<<<xgrid, 256, 0, stream>>>(h1, Xt, N_, D_);
    apl_gemm<<<ggrid, 256, 0, stream>>>(Xt, ph1, vh1t, hbar, 0);
    apl_gemm<<<ggrid, 256, 0, stream>>>(Xt, pz1, vz1t, zbuf, 1);
    rec_pass1<<<256, 256, 0, stream>>>(zbuf, hbar, Aprod, Hend);
    rec_pass2<<<16, 256, 0, stream>>>(Aprod, Hend, Hstart);
    rec_pass3<<<256, 256, 0, stream>>>(zbuf, hbar, Hstart, h2);
    maxabs_norm<<<N_ / 8, 256, 0, stream>>>(h2);

    // ---- Output projection
    transpose_f32<<<xgrid, 256, 0, stream>>>(h2, Xt, N_, D_);
    apl_gemm<<<ggrid, 256, 0, stream>>>(Xt, po, vot, out, 0);
}
